// CombinedLoss_28802050687819
// MI455X (gfx1250) — compile-verified, matrix operand order checked by reading, not run on hardware
//
#include <hip/hip_runtime.h>
#include <hip/hip_bf16.h>

typedef __attribute__((ext_vector_type(16))) _Float16 v16h;
typedef __attribute__((ext_vector_type(8)))  float    v8f;

#define VOL (128*128*128)

__device__ __forceinline__ float wave_sum(float v) {
#pragma unroll
  for (int o = 16; o > 0; o >>= 1) v += __shfl_xor(v, o, 32);
  return v;
}

// ---------------- workspace layout (bytes) ----------------
// [0, 512)        : 128 floats of accumulators:
//   F[0..5]   inter[b][c], F[6..11] psum[b][c], F[12..17] ohsum[b][c]
//   F[24 + (s*2+b)*12 + k] : contrastive sums, k=0..8 -> S[i*3+j], k=9..11 -> D[i]
//   F[96 + (s*2+b)*3 + i]  : class counts
// [512, 37888)    : 9344 ints: masks (scale0 @0 len 8192, scale1 @8192 len 1024, scale2 @9216 len 128)
// [37888, ...)    : f16 normalized features, transposed (N-major, K padded to 64)
//   scale0 @0 (2*4096*64), scale1 @524288 (2*512*64), scale2 @589824 (2*64*64)

__global__ void zero_ws_kernel(float* __restrict__ F) {
  int i = threadIdx.x;
  if (i < 128) F[i] = 0.0f;
}

// ---------------- dice partials ----------------
__global__ void dice_kernel(const float* __restrict__ pred,
                            const int* __restrict__ target,
                            float* __restrict__ F) {
  const int b = blockIdx.y;
  const float* pb = pred + (size_t)b * 3 * VOL;
  const int*   tb = target + (size_t)b * VOL;
  int idx = blockIdx.x * blockDim.x + threadIdx.x;
  int stride = gridDim.x * blockDim.x;
  float i0=0,i1=0,i2=0, p0s=0,p1s=0,p2s=0, o0=0,o1=0,o2=0;
  for (int v = idx; v < VOL; v += stride) {
    float x0 = pb[v];
    float x1 = pb[v + VOL];
    float x2 = pb[v + 2*VOL];
    float mx = fmaxf(x0, fmaxf(x1, x2));
    float e0 = __expf(x0-mx), e1 = __expf(x1-mx), e2 = __expf(x2-mx);
    float inv = 1.0f / (e0 + e1 + e2);
    float p0 = e0*inv, p1 = e1*inv, p2 = e2*inv;
    int t = tb[v];
    t = max(0, min(2, t));
    float t0 = (t==0)?1.0f:0.0f, t1 = (t==1)?1.0f:0.0f, t2 = (t==2)?1.0f:0.0f;
    p0s += p0; p1s += p1; p2s += p2;
    o0 += t0;  o1 += t1;  o2 += t2;
    i0 += t0*p0; i1 += t1*p1; i2 += t2*p2;
  }
  bool lead = (threadIdx.x & 31) == 0;
  float s;
  s = wave_sum(i0);  if (lead) atomicAdd(&F[b*3+0], s);
  s = wave_sum(i1);  if (lead) atomicAdd(&F[b*3+1], s);
  s = wave_sum(i2);  if (lead) atomicAdd(&F[b*3+2], s);
  s = wave_sum(p0s); if (lead) atomicAdd(&F[6+b*3+0], s);
  s = wave_sum(p1s); if (lead) atomicAdd(&F[6+b*3+1], s);
  s = wave_sum(p2s); if (lead) atomicAdd(&F[6+b*3+2], s);
  s = wave_sum(o0);  if (lead) atomicAdd(&F[12+b*3+0], s);
  s = wave_sum(o1);  if (lead) atomicAdd(&F[12+b*3+1], s);
  s = wave_sum(o2);  if (lead) atomicAdd(&F[12+b*3+2], s);
}

// ---------------- per-scale prep: mask + L2-normalize -> f16 (N-major, K=64) ----------------
__global__ void prep_kernel(const float* __restrict__ feat, const int* __restrict__ target,
                            _Float16* __restrict__ H, int* __restrict__ mask,
                            float* __restrict__ counts, int N, int dim, int stride) {
  int idx = blockIdx.x * blockDim.x + threadIdx.x;
  if (idx >= 2*N) return;
  int b = idx / N;
  int n = idx - b*N;
  int x = n % dim;
  int y = (n / dim) % dim;
  int z = n / (dim*dim);
  int t = target[(((size_t)b*128 + (size_t)z*stride)*128 + (size_t)y*stride)*128 + (size_t)x*stride];
  t = max(0, min(2, t));
  mask[idx] = t;
  atomicAdd(&counts[b*3 + t], 1.0f);

  const float* fp = feat + (size_t)b * 48 * N + n;
  float ss = 0.0f;
#pragma unroll 4
  for (int c = 0; c < 48; ++c) { float v = fp[(size_t)c*N]; ss += v*v; }
  float inv = 1.0f / fmaxf(sqrtf(ss), 1e-12f);
  _Float16* o = H + (size_t)idx * 64;
#pragma unroll 4
  for (int c = 0; c < 48; ++c) o[c] = (_Float16)(fp[(size_t)c*N] * inv);
#pragma unroll
  for (int c = 48; c < 64; ++c) o[c] = (_Float16)0.0f;
}

// Load one 16x32 f16 operand fragment (A and B share the K-along-registers layout).
// Memory layout: fT[col*64 + k]. ISA 16-bit A layout:
//   lane (0..31): col = lane%16 ; k base = (lane<16 ? 0 : 8)
//   VGPR v in 0..3: k = kb + 2v {+0,+1} ; v in 4..7: k = 16 + kb + 2(v-4) {+0,+1}
__device__ __forceinline__ v16h load_tile(const _Float16* __restrict__ fT,
                                          int col_base, int koff, int lane) {
  int col = col_base + (lane & 15);
  int kb  = (lane < 16) ? 0 : 8;
  const _Float16* p = fT + (size_t)col * 64 + koff;
  v16h r;
#pragma unroll
  for (int v = 0; v < 8; ++v) {
    int kl = (v < 4) ? (kb + 2*v) : (16 + kb + 2*(v-4));
    r[2*v]   = p[kl];
    r[2*v+1] = p[kl+1];
  }
  return r;
}

// ---------------- contrastive Gram-matrix kernel (WMMA f16 -> f32) ----------------
__global__ void __launch_bounds__(128) contr_gemm_kernel(
    const _Float16* __restrict__ H, const int* __restrict__ mask,
    float* __restrict__ sums, int N) {
  int lane = threadIdx.x & 31;
  int wave = blockIdx.x * (blockDim.x >> 5) + (threadIdx.x >> 5);
  int rowsPerBatch = N >> 4;
  int b  = wave / rowsPerBatch;
  int tr = wave - b * rowsPerBatch;
  int M0 = tr << 4;
  const _Float16* fb = H + (size_t)b * N * 64;
  const int* mb = mask + b * N;
  float* sb = sums + b * 12;

  // A fragments live in VGPRs for the whole row strip (K = 48 padded to 64)
  v16h a0 = load_tile(fb, M0, 0,  lane);
  v16h a1 = load_tile(fb, M0, 32, lane);

  int rowOff = (lane < 16) ? 0 : 8;   // C/D layout: VGPR r holds M = r + rowOff
  int mi[8];
#pragma unroll
  for (int r = 0; r < 8; ++r) mi[r] = mb[M0 + r + rowOff];

  float acc[9] = {0,0,0,0,0,0,0,0,0};
  float d0 = 0.0f, d1 = 0.0f, d2 = 0.0f;

  for (int N0 = 0; N0 < N; N0 += 16) {
    v16h b0 = load_tile(fb, N0, 0,  lane);
    v16h b1 = load_tile(fb, N0, 32, lane);
    v8f c = {};
    c = __builtin_amdgcn_wmma_f32_16x16x32_f16(false, a0, false, b0, (short)0, c, false, false);
    c = __builtin_amdgcn_wmma_f32_16x16x32_f16(false, a1, false, b1, (short)0, c, false, false);

    int n_g = N0 + (lane & 15);
    int j = mb[n_g];                 // column class: uniform per lane per tile
    float s0 = 0.0f, s1 = 0.0f, s2 = 0.0f;
#pragma unroll
    for (int r = 0; r < 8; ++r) {
      float e  = __expf(c[r] * 10.0f);        // exp(dot / TEMP), TEMP = 0.1
      int  cls = mi[r];                        // row class: loop-invariant per lane
      bool dg  = (M0 + r + rowOff) == n_g;     // diagonal element
      float ed = dg ? e : 0.0f;
      s0 += (cls==0) ? e : 0.0f;
      s1 += (cls==1) ? e : 0.0f;
      s2 += (cls==2) ? e : 0.0f;
      d0 += (cls==0) ? ed : 0.0f;
      d1 += (cls==1) ? ed : 0.0f;
      d2 += (cls==2) ? ed : 0.0f;
    }
    float w0 = (j==0)?1.0f:0.0f, w1 = (j==1)?1.0f:0.0f, w2 = (j==2)?1.0f:0.0f;
    acc[0] += s0*w0; acc[1] += s0*w1; acc[2] += s0*w2;
    acc[3] += s1*w0; acc[4] += s1*w1; acc[5] += s1*w2;
    acc[6] += s2*w0; acc[7] += s2*w1; acc[8] += s2*w2;
  }

  bool lead = (lane == 0);
#pragma unroll
  for (int k = 0; k < 9; ++k) {
    float s = wave_sum(acc[k]);
    if (lead) atomicAdd(&sb[k], s);
  }
  float s;
  s = wave_sum(d0); if (lead) atomicAdd(&sb[9],  s);
  s = wave_sum(d1); if (lead) atomicAdd(&sb[10], s);
  s = wave_sum(d2); if (lead) atomicAdd(&sb[11], s);
}

// ---------------- finalize ----------------
__device__ __forceinline__ float pair_term(float sumEW, float cnt, bool positive) {
  float mean = sumEW / fmaxf(cnt, 1.0f);
  float val = positive ? -__logf(mean + 1e-6f) : __logf(1.0f + mean);
  return (cnt > 0.0f) ? val : 0.0f;
}

__global__ void finalize_kernel(const float* __restrict__ F,
                                const float* __restrict__ logits,
                                const int* __restrict__ labels,
                                float* __restrict__ out) {
  if (threadIdx.x != 0) return;
  // dice
  float dsum = 0.0f;
#pragma unroll
  for (int b = 0; b < 2; ++b)
#pragma unroll
    for (int c = 1; c < 3; ++c) {
      float inter = F[b*3+c], ps = F[6+b*3+c], os = F[12+b*3+c];
      dsum += (2.0f*inter + 1e-5f) / (ps + os + 1e-5f);
    }
  float dice = 1.0f - dsum * 0.25f;
  // focal
  float fsum = 0.0f;
#pragma unroll
  for (int k = 0; k < 6; ++k) {
    float l  = logits[k];
    float pt = 1.0f / (1.0f + __expf(-l));
    if (labels[k] != 1) pt = 1.0f - pt;
    float om = 1.0f - pt;
    fsum += -0.25f * om * om * __logf(pt + 1e-6f);
  }
  float focal = fsum * (1.0f / 6.0f);
  // contrastive
  const float sw0 = 1.0f, sw1 = 0.8f, sw2 = 0.6f;
  float per_scale[3];
#pragma unroll
  for (int s = 0; s < 3; ++s) {
    float bs = 0.0f;
#pragma unroll
    for (int b = 0; b < 2; ++b) {
      const float* S = F + 24 + (s*2+b)*12;
      const float* C = F + 96 + (s*2+b)*3;
      float c0 = C[0], c1 = C[1], c2 = C[2];
      bs += pair_term(S[4] - S[10], c1*c1 - c1, true);   // (w1,w1) offdiag
      bs += pair_term(S[8] - S[11], c2*c2 - c2, true);   // (w2,w2) offdiag
      bs += pair_term(S[5], c1*c2, false);               // (w1,w2)
      bs += pair_term(S[1], c0*c1, false);               // (w0,w1)
      bs += pair_term(S[2], c0*c2, false);               // (w0,w2)
    }
    per_scale[s] = bs * 0.5f;                            // mean over batch
  }
  float contr = sw0*per_scale[0] + sw1*per_scale[1] + sw2*per_scale[2];

  out[0] = dice + focal + 0.1f * contr;
  out[1] = dice;
  out[2] = focal;
  out[3] = contr;
}

extern "C" void kernel_launch(void* const* d_in, const int* in_sizes, int n_in,
                              void* d_out, int out_size, void* d_ws, size_t ws_size,
                              hipStream_t stream) {
  const float* pred   = (const float*)d_in[0];
  const int*   target = (const int*)d_in[1];
  const float* fmap0  = (const float*)d_in[2];
  const float* fmap1  = (const float*)d_in[3];
  const float* fmap2  = (const float*)d_in[4];
  const float* logits = (const float*)d_in[5];
  const int*   labels = (const int*)d_in[6];
  float* out = (float*)d_out;

  float*    F = (float*)d_ws;
  int*      M = (int*)((char*)d_ws + 512);
  _Float16* H = (_Float16*)((char*)d_ws + 37888);

  zero_ws_kernel<<<1, 128, 0, stream>>>(F);
  dice_kernel<<<dim3(512, 2), 256, 0, stream>>>(pred, target, F);

  const float* feats[3]   = {fmap0, fmap1, fmap2};
  const int    Ns[3]      = {4096, 512, 64};
  const int    dims[3]    = {16, 8, 4};
  const int    strides[3] = {8, 16, 32};
  const int    mOff[3]    = {0, 8192, 9216};
  const size_t hOff[3]    = {0, 524288, 589824};

  for (int s = 0; s < 3; ++s) {
    int n2 = 2 * Ns[s];
    int blocks = (n2 + 255) / 256;
    prep_kernel<<<blocks, 256, 0, stream>>>(feats[s], target, H + hOff[s], M + mOff[s],
                                            F + 96 + s*6, Ns[s], dims[s], strides[s]);
  }
  for (int s = 0; s < 3; ++s) {
    int totalWaves = 2 * (Ns[s] >> 4);   // 512, 64, 8 -> all divisible by 4
    int blocks = totalWaves / 4;
    contr_gemm_kernel<<<blocks, 128, 0, stream>>>(H + hOff[s], M + mOff[s],
                                                  F + 24 + s*24, Ns[s]);
  }
  finalize_kernel<<<1, 32, 0, stream>>>(F, logits, labels, out);
}
